// GroupedQueryAttention_78615081386062
// MI455X (gfx1250) — matrix operand order repaired and checked by reading, not compiled, not run on hardware
//
#include <hip/hip_runtime.h>
#include <hip/hip_bf16.h>

// ---------------------------------------------------------------------------
// GQA forward for MI455X (gfx1250, wave32, WMMA).
// bf16 operands + fp32 accumulation via v_wmma_f32_16x16x32_bf16.
// GEMMs: cooperative async global->LDS panel copies (ASYNCcnt, double
// buffered) + ds_load_b128 fragment reads. This revision builds all four
// B fragments before issuing the four WMMAs back-to-back, eliminating the
// WMMA->VALU WAR hazard v_nops seen in round 3.
// ---------------------------------------------------------------------------

#define S_LEN   2048
#define DM      2048
#define NH      32
#define NKV     8
#define HD      64
#define KVD     (NKV * HD)     // 512
#define NTOT    (DM + 2 * KVD) // 3072 fused QKV output width

typedef __attribute__((ext_vector_type(16))) __bf16 v16bf;
typedef __attribute__((ext_vector_type(8)))  float  v8f;

static __device__ __forceinline__ v8f wmma_bf16(v16bf a, v16bf b, v8f c) {
    return __builtin_amdgcn_wmma_f32_16x16x32_bf16(
        /*neg_a=*/false, a, /*neg_b=*/false, b,
        /*c_mod=*/(short)0, c, /*reuse_a=*/false, /*reuse_b=*/false);
}

// ---- CDNA5 async global->LDS copy (ASYNCcnt) ------------------------------
static __device__ __forceinline__ void async_copy_b128(unsigned lds_byte, const void* gaddr) {
    asm volatile("global_load_async_to_lds_b128 %0, %1, off"
                 :: "v"(lds_byte), "v"(gaddr)
                 : "memory");
}
#define S_WAIT_ASYNCCNT(n) asm volatile("s_wait_asynccnt " #n ::: "memory")

static __device__ __forceinline__ unsigned lds_addr_of(const void* p) {
    // addrspace(3) flat pointers carry the LDS byte offset in the low 32 bits
    return (unsigned)(unsigned long long)p;
}

union BFrag { v16bf v; uint4 q[2]; };

// 16 contiguous bf16 -> fragment
static __device__ __forceinline__ v16bf load16_contig(const __bf16* p) {
    BFrag f;
    f.q[0] = *(const uint4*)(p);
    f.q[1] = *(const uint4*)(p + 8);
    return f.v;
}
// 16 contiguous bf16 at p (A-fragment layout)
static __device__ __forceinline__ v16bf load16_2runs(const __bf16* p) {
    BFrag f;
    f.q[0] = *(const uint4*)(p);
    f.q[1] = *(const uint4*)(p + 8);
    return f.v;
}
static __device__ __forceinline__ void cvt4(v16bf& d, int base, float4 f) {
    d[base + 0] = (__bf16)f.x;
    d[base + 1] = (__bf16)f.y;
    d[base + 2] = (__bf16)f.z;
    d[base + 3] = (__bf16)f.w;
}
// 16 contiguous fp32 -> bf16 fragment (B layout)
static __device__ __forceinline__ v16bf frag16_f32(const float* p) {
    float4 f0 = *(const float4*)(p);
    float4 f1 = *(const float4*)(p + 4);
    float4 f2 = *(const float4*)(p + 8);
    float4 f3 = *(const float4*)(p + 12);
    v16bf v;
    cvt4(v, 0, f0); cvt4(v, 4, f1); cvt4(v, 8, f2); cvt4(v, 12, f3);
    return v;
}
// 16 contiguous fp32 at p -> bf16 fragment (A layout)
static __device__ __forceinline__ v16bf fragA_f32(const float* p) {
    float4 f0 = *(const float4*)(p);
    float4 f1 = *(const float4*)(p + 4);
    float4 f2 = *(const float4*)(p + 8);
    float4 f3 = *(const float4*)(p + 12);
    v16bf v;
    cvt4(v, 0, f0); cvt4(v, 4, f1); cvt4(v, 8, f2); cvt4(v, 12, f3);
    return v;
}

#define APITCH 36   // fp32 LDS row pitch (bank-conflict-free, 16B aligned)
#define BPITCH 36
#define AOPITCH 40  // bf16 LDS row pitch

// ---------------------------------------------------------------------------
// Kernel 1: fused QKV projection.  Block = 8 waves -> 128x64 macro-tile.
// Per k-step: async-copy A(128x32 fp32) + B(64x32 fp32) panels to LDS
// (double buffered); build A + all 4 B fragments, then 4 back-to-back WMMAs.
// ---------------------------------------------------------------------------
__global__ void qkv_gemm_kernel(const float* __restrict__ x,
                                const float* __restrict__ Wq,
                                const float* __restrict__ Wk,
                                const float* __restrict__ Wv,
                                float* __restrict__ Qf,
                                float* __restrict__ Kf,
                                float* __restrict__ Vf) {
    __shared__ alignas(16) float AS[2][128 * APITCH];  // 36 KB
    __shared__ alignas(16) float BS[2][64 * BPITCH];   // 18 KB

    const int tid  = threadIdx.x;
    const int lane = tid & 31;
    const int w    = tid >> 5;
    const int nstrips = NTOT / 64;                 // 48
    const int m0blk = (int)(blockIdx.x / nstrips) * 128;
    const int n0    = (int)(blockIdx.x % nstrips) * 64;
    const int nl = lane & 15;
    const int hi = lane >> 4;

    const float* Wbase; int nrow0;
    if (n0 < DM)            { Wbase = Wq; nrow0 = n0; }
    else if (n0 < DM + KVD) { Wbase = Wk; nrow0 = n0 - DM; }
    else                    { Wbase = Wv; nrow0 = n0 - DM - KVD; }

    const unsigned aL[2] = { lds_addr_of(&AS[0][0]), lds_addr_of(&AS[1][0]) };
    const unsigned bL[2] = { lds_addr_of(&BS[0][0]), lds_addr_of(&BS[1][0]) };

    auto fill = [&](int buf, int k0) {
#pragma unroll
        for (int i = 0; i < 4; ++i) {              // A: 128 rows x 8 chunks
            int id  = tid + 256 * i;
            int row = id >> 3;
            int c4  = (id & 7) * 4;
            async_copy_b128(aL[buf] + (unsigned)(row * APITCH + c4) * 4,
                            x + (size_t)(m0blk + row) * DM + k0 + c4);
        }
#pragma unroll
        for (int i = 0; i < 2; ++i) {              // B: 64 rows x 8 chunks
            int id  = tid + 256 * i;
            int row = id >> 3;
            int c4  = (id & 7) * 4;
            async_copy_b128(bL[buf] + (unsigned)(row * BPITCH + c4) * 4,
                            Wbase + (size_t)(nrow0 + row) * DM + k0 + c4);
        }
    };

    v8f acc0 = {}, acc1 = {}, acc2 = {}, acc3 = {};

    const int STEPS = DM / 32;                     // 64
    fill(0, 0);
    for (int i = 0; i < STEPS; ++i) {
        const int cur = i & 1;
        if (i + 1 < STEPS) {
            fill(cur ^ 1, (i + 1) * 32);
            S_WAIT_ASYNCCNT(6);                    // prev panel's 6 ops done
        } else {
            S_WAIT_ASYNCCNT(0);
        }
        __syncthreads();

        // build all fragments first (distinct regs), then WMMAs back-to-back
        v16bf a  = fragA_f32(&AS[cur][(w * 16 + nl) * APITCH + hi * 16]);
        v16bf b0 = frag16_f32(&BS[cur][( 0 + nl) * BPITCH + hi * 16]);
        v16bf b1 = frag16_f32(&BS[cur][(16 + nl) * BPITCH + hi * 16]);
        v16bf b2 = frag16_f32(&BS[cur][(32 + nl) * BPITCH + hi * 16]);
        v16bf b3 = frag16_f32(&BS[cur][(48 + nl) * BPITCH + hi * 16]);
        acc0 = wmma_bf16(a, b0, acc0);
        acc1 = wmma_bf16(a, b1, acc1);
        acc2 = wmma_bf16(a, b2, acc2);
        acc3 = wmma_bf16(a, b3, acc3);

        __syncthreads();                           // all waves done reading
    }

    const int m0 = m0blk + w * 16;
    v8f accs[4] = {acc0, acc1, acc2, acc3};
#pragma unroll
    for (int t = 0; t < 4; ++t) {
        int n = n0 + t * 16 + nl;
        float* out; int col; int ld;
        if (n < DM)            { out = Qf; col = n;            ld = DM; }
        else if (n < DM + KVD) { out = Kf; col = n - DM;       ld = KVD; }
        else                   { out = Vf; col = n - DM - KVD; ld = KVD; }
#pragma unroll
        for (int r = 0; r < 8; ++r) {
            int m = m0 + r + 8 * hi;
            out[(size_t)m * ld + col] = accs[t][r];
        }
    }
}

// ---------------------------------------------------------------------------
// Kernel 2: RoPE on Q/K + pack to bf16 head-major; V packed transposed.
// ---------------------------------------------------------------------------
__global__ void rope_pack_kernel(const float* __restrict__ Qf,
                                 const float* __restrict__ Kf,
                                 const float* __restrict__ Vf,
                                 __bf16* __restrict__ Qh,
                                 __bf16* __restrict__ Kh,
                                 __bf16* __restrict__ Vt) {
    const int QN = NH  * S_LEN * (HD / 2);
    const int KN = NKV * S_LEN * (HD / 2);
    const int VN = NKV * S_LEN * (HD / 2);
    int id = (int)(blockIdx.x * blockDim.x + threadIdx.x);
    if (id >= QN + KN + VN) return;

    if (id < QN) {
        int d = id & 31, s = (id >> 5) & (S_LEN - 1), h = id >> 16;
        float freq = __powf(10000.0f, -(float)(2 * d) * (1.0f / (float)HD));
        float ang  = (float)s * freq;
        float c, sn; __sincosf(ang, &sn, &c);
        float q1 = Qf[(size_t)s * DM + h * HD + d];
        float q2 = Qf[(size_t)s * DM + h * HD + d + 32];
        __bf16* dst = Qh + ((size_t)h * S_LEN + s) * HD + d;
        dst[0]  = (__bf16)(q1 * c - q2 * sn);
        dst[32] = (__bf16)(q2 * c + q1 * sn);
    } else if (id < QN + KN) {
        int id2 = id - QN;
        int d = id2 & 31, s = (id2 >> 5) & (S_LEN - 1), kv = id2 >> 16;
        float freq = __powf(10000.0f, -(float)(2 * d) * (1.0f / (float)HD));
        float ang  = (float)s * freq;
        float c, sn; __sincosf(ang, &sn, &c);
        float k1 = Kf[(size_t)s * KVD + kv * HD + d];
        float k2 = Kf[(size_t)s * KVD + kv * HD + d + 32];
        __bf16* dst = Kh + ((size_t)kv * S_LEN + s) * HD + d;
        dst[0]  = (__bf16)(k1 * c - k2 * sn);
        dst[32] = (__bf16)(k2 * c + k1 * sn);
    } else {
        int id3 = id - QN - KN;
        int d = id3 & 31, s = (id3 >> 5) & (S_LEN - 1), kv = id3 >> 16;
        float v1 = Vf[(size_t)s * KVD + kv * HD + d];
        float v2 = Vf[(size_t)s * KVD + kv * HD + d + 32];
        Vt[((size_t)kv * HD + d)      * S_LEN + s] = (__bf16)v1;
        Vt[((size_t)kv * HD + d + 32) * S_LEN + s] = (__bf16)v2;
    }
}

// ---------------------------------------------------------------------------
// Kernel 3: flash-style attention. One wave per (head, 16-query tile).
// K fragments ping-pong in NAMED struct members (SROA-safe), manual
// unroll-by-2 over 32-key chunks.
// ---------------------------------------------------------------------------
struct KF { v16bf k00, k10, k01, k11; };   // [key-tile 0/1][k-dim half 0/1]

__global__ void attn_kernel(const __bf16* __restrict__ Qh,
                            const __bf16* __restrict__ Kh,
                            const __bf16* __restrict__ Vt,
                            __bf16* __restrict__ Ao) {
    __shared__ __bf16 lds[8][16][32];   // per-wave 16x32 P tile

    const int lane = threadIdx.x & 31;
    const int w    = threadIdx.x >> 5;
    const int wave = (int)((blockIdx.x * blockDim.x + threadIdx.x) >> 5);
    const int qt = wave & 127;
    const int h  = wave >> 7;
    const int kv = h >> 2;
    const int nl = lane & 15;
    const int hi = lane >> 4;
    const float scale = 0.125f;  // 1/sqrt(64)

    const __bf16* qrow = Qh + ((size_t)h * S_LEN + qt * 16 + nl) * HD;
    v16bf qa0 = load16_2runs(qrow + hi * 16);
    v16bf qa1 = load16_2runs(qrow + 32 + hi * 16);

    const __bf16* kbase0 = Kh + ((size_t)kv * S_LEN + nl)      * HD + hi * 16;
    const __bf16* kbase1 = Kh + ((size_t)kv * S_LEN + 16 + nl) * HD + hi * 16;
    const __bf16* vbase  = Vt + ((size_t)kv * HD + nl) * S_LEN + hi * 16;

    v8f o[4] = {v8f{}, v8f{}, v8f{}, v8f{}};
    float mi[8], li[8];
#pragma unroll
    for (int r = 0; r < 8; ++r) { mi[r] = -1e30f; li[r] = 0.0f; }

    auto load_kf = [&](KF& f, int n0) {
        const size_t off = (size_t)n0 * HD;
        f.k00 = load16_contig(kbase0 + off);
        f.k10 = load16_contig(kbase1 + off);
        f.k01 = load16_contig(kbase0 + off + 32);
        f.k11 = load16_contig(kbase1 + off + 32);
    };

    auto chunk = [&](int n0, const KF& kf, KF& kfn, bool preload) {
        // ---- scores ----
        v8f s0 = {}, s1 = {};
        s0 = wmma_bf16(qa0, kf.k00, s0);
        s1 = wmma_bf16(qa0, kf.k10, s1);
        s0 = wmma_bf16(qa1, kf.k01, s0);
        s1 = wmma_bf16(qa1, kf.k11, s1);

        if (preload) load_kf(kfn, n0 + 32);        // covered by softmax below

        v16bf vb0 = load16_contig(vbase + (size_t)0  * S_LEN + n0);
        v16bf vb1 = load16_contig(vbase + (size_t)16 * S_LEN + n0);
        v16bf vb2 = load16_contig(vbase + (size_t)32 * S_LEN + n0);
        v16bf vb3 = load16_contig(vbase + (size_t)48 * S_LEN + n0);

        // ---- online softmax ----
        float p0v[8], p1v[8], alpha[8];
#pragma unroll
        for (int r = 0; r < 8; ++r) {
            float v0 = s0[r] * scale;
            float v1 = s1[r] * scale;
            float m  = fmaxf(v0, v1);
#pragma unroll
            for (int off = 1; off < 16; off <<= 1)
                m = fmaxf(m, __shfl_xor(m, off, 32));
            float mn = fmaxf(mi[r], m);
            alpha[r] = __expf(mi[r] - mn);
            mi[r]    = mn;
            p0v[r]   = __expf(v0 - mn);
            p1v[r]   = __expf(v1 - mn);
            float sum = p0v[r] + p1v[r];
#pragma unroll
            for (int off = 1; off < 16; off <<= 1)
                sum += __shfl_xor(sum, off, 32);
            li[r] = li[r] * alpha[r] + sum;
        }

#pragma unroll
        for (int t = 0; t < 4; ++t)
#pragma unroll
            for (int r = 0; r < 8; ++r) o[t][r] *= alpha[r];

        // ---- P transpose through LDS ----
        __syncthreads();
#pragma unroll
        for (int r = 0; r < 8; ++r) {
            lds[w][r + 8 * hi][nl]      = (__bf16)p0v[r];
            lds[w][r + 8 * hi][16 + nl] = (__bf16)p1v[r];
        }
        __syncthreads();
        v16bf pa = load16_2runs(&lds[w][nl][hi * 16]);

        // ---- P @ V (distinct preloaded B fragments; back-to-back) ----
        o[0] = wmma_bf16(pa, vb0, o[0]);
        o[1] = wmma_bf16(pa, vb1, o[1]);
        o[2] = wmma_bf16(pa, vb2, o[2]);
        o[3] = wmma_bf16(pa, vb3, o[3]);
    };

    const int CH = S_LEN / 32;   // 64 (even)
    KF kfA, kfB;
    load_kf(kfA, 0);
    for (int c = 0; c < CH; c += 2) {
        chunk(c * 32,       kfA, kfB, true);
        chunk((c + 1) * 32, kfB, kfA, (c + 2) < CH);
    }

#pragma unroll
    for (int t = 0; t < 4; ++t) {
#pragma unroll
        for (int r = 0; r < 8; ++r) {
            float inv = 1.0f / li[r];
            int srow = qt * 16 + r + 8 * hi;
            Ao[(size_t)srow * DM + h * HD + t * 16 + nl] = (__bf16)(o[t][r] * inv);
        }
    }
}

// ---------------------------------------------------------------------------
// Kernel 4: output projection.  Block = 8 waves -> 128x64 macro-tile.
// A panel (bf16) + B panel (fp32) async-copied to double-buffered LDS.
// ---------------------------------------------------------------------------
__global__ void out_gemm_kernel(const __bf16* __restrict__ Ao,
                                const float* __restrict__ Wo,
                                float* __restrict__ out) {
    __shared__ alignas(16) __bf16 ASH[2][128 * AOPITCH]; // 20 KB
    __shared__ alignas(16) float  BS[2][64 * BPITCH];    // 18 KB

    const int tid  = threadIdx.x;
    const int lane = tid & 31;
    const int w    = tid >> 5;
    const int nstrips = DM / 64;                   // 32
    const int m0blk = (int)(blockIdx.x / nstrips) * 128;
    const int n0    = (int)(blockIdx.x % nstrips) * 64;
    const int nl = lane & 15;
    const int hi = lane >> 4;

    const unsigned aL[2] = { lds_addr_of(&ASH[0][0]), lds_addr_of(&ASH[1][0]) };
    const unsigned bL[2] = { lds_addr_of(&BS[0][0]),  lds_addr_of(&BS[1][0])  };

    auto fill = [&](int buf, int k0) {
#pragma unroll
        for (int i = 0; i < 2; ++i) {              // A: 128 rows x 4 chunks (bf16)
            int id  = tid + 256 * i;
            int row = id >> 2;
            int c8  = (id & 3) * 8;
            async_copy_b128(aL[buf] + (unsigned)(row * AOPITCH + c8) * 2,
                            Ao + (size_t)(m0blk + row) * DM + k0 + c8);
        }
#pragma unroll
        for (int i = 0; i < 2; ++i) {              // B: 64 rows x 8 chunks (fp32)
            int id  = tid + 256 * i;
            int row = id >> 3;
            int c4  = (id & 7) * 4;
            async_copy_b128(bL[buf] + (unsigned)(row * BPITCH + c4) * 4,
                            Wo + (size_t)(n0 + row) * DM + k0 + c4);
        }
    };

    v8f acc0 = {}, acc1 = {}, acc2 = {}, acc3 = {};

    const int STEPS = DM / 32;
    fill(0, 0);
    for (int i = 0; i < STEPS; ++i) {
        const int cur = i & 1;
        if (i + 1 < STEPS) {
            fill(cur ^ 1, (i + 1) * 32);
            S_WAIT_ASYNCCNT(4);
        } else {
            S_WAIT_ASYNCCNT(0);
        }
        __syncthreads();

        v16bf a  = load16_2runs(&ASH[cur][(w * 16 + nl) * AOPITCH + hi * 16]);
        v16bf b0 = frag16_f32(&BS[cur][( 0 + nl) * BPITCH + hi * 16]);
        v16bf b1 = frag16_f32(&BS[cur][(16 + nl) * BPITCH + hi * 16]);
        v16bf b2 = frag16_f32(&BS[cur][(32 + nl) * BPITCH + hi * 16]);
        v16bf b3 = frag16_f32(&BS[cur][(48 + nl) * BPITCH + hi * 16]);
        acc0 = wmma_bf16(a, b0, acc0);
        acc1 = wmma_bf16(a, b1, acc1);
        acc2 = wmma_bf16(a, b2, acc2);
        acc3 = wmma_bf16(a, b3, acc3);

        __syncthreads();
    }

    const int m0 = m0blk + w * 16;
    v8f accs[4] = {acc0, acc1, acc2, acc3};
#pragma unroll
    for (int t = 0; t < 4; ++t) {
        int col = n0 + t * 16 + nl;
#pragma unroll
        for (int r = 0; r < 8; ++r) {
            int m = m0 + r + 8 * hi;
            out[(size_t)m * DM + col] = accs[t][r];
        }
    }
}

// ---------------------------------------------------------------------------
extern "C" void kernel_launch(void* const* d_in, const int* in_sizes, int n_in,
                              void* d_out, int out_size, void* d_ws, size_t ws_size,
                              hipStream_t stream) {
    const float* x  = (const float*)d_in[0];
    const float* Wq = (const float*)d_in[1];
    const float* Wk = (const float*)d_in[2];
    const float* Wv = (const float*)d_in[3];
    const float* Wo = (const float*)d_in[4];
    float* out = (float*)d_out;

    char* p = (char*)d_ws;
    auto carve = [&](size_t bytes) {
        char* r = p;
        p += (bytes + 255) & ~(size_t)255;
        return r;
    };
    float*  Qf = (float*)carve((size_t)S_LEN * DM  * sizeof(float));   // 16 MB
    float*  Kf = (float*)carve((size_t)S_LEN * KVD * sizeof(float));   //  4 MB
    float*  Vf = (float*)carve((size_t)S_LEN * KVD * sizeof(float));   //  4 MB
    __bf16* Qh = (__bf16*)carve((size_t)NH  * S_LEN * HD * 2);         //  8 MB
    __bf16* Kh = (__bf16*)carve((size_t)NKV * S_LEN * HD * 2);         //  2 MB
    __bf16* Vt = (__bf16*)carve((size_t)NKV * HD * S_LEN * 2);         //  2 MB
    __bf16* Ao = (__bf16*)carve((size_t)S_LEN * DM * 2);               //  8 MB
    (void)ws_size; (void)in_sizes; (void)n_in; (void)out_size;

    {   // K1: fused QKV GEMM — 16 m-blocks x 48 strips = 768 blocks
        int blocks = (DM / 128) * (NTOT / 64);
        qkv_gemm_kernel<<<blocks, 256, 0, stream>>>(x, Wq, Wk, Wv, Qf, Kf, Vf);
    }
    {   // K2: RoPE + pack
        int total = (NH + NKV + NKV) * S_LEN * (HD / 2);
        rope_pack_kernel<<<(total + 255) / 256, 256, 0, stream>>>(Qf, Kf, Vf, Qh, Kh, Vt);
    }
    {   // K3: attention — 32 heads x 128 q-tiles = 4096 waves
        int waves = NH * (S_LEN / 16);
        attn_kernel<<<(waves * 32) / 256, 256, 0, stream>>>(Qh, Kh, Vt, Ao);
    }
    {   // K4: output projection — 16 x 32 = 512 blocks
        int blocks = (DM / 128) * (DM / 64);
        out_gemm_kernel<<<blocks, 256, 0, stream>>>(Ao, Wo, out);
    }
}
